// NeuralRNN_50603304682319
// MI455X (gfx1250) — matrix operand order repaired and checked by reading, not compiled, not run on hardware
//
#include <hip/hip_runtime.h>
#include <hip/hip_bf16.h>

// Elman RNN on gfx1250 (MI455X), wave32 + WMMA bf16.
//   Kernel 1 (xh_gemm):   xp = x @ W_xh^T + b_xh, written into d_out's h_seq slot.
//   Kernel 2 (recurrence): single persistent WG (512 thr / 16 waves) on one WGP.
//     W_hh fragments hoisted to VGPRs (invariant over 2048 steps), h in LDS (bf16),
//     xp reads software-pipelined one step ahead, tanh via v_exp_f32 + v_rcp_f32.

typedef __attribute__((ext_vector_type(16))) __bf16 v16bf;
typedef __attribute__((ext_vector_type(8)))  __bf16 v8bf;
typedef __attribute__((ext_vector_type(8)))  float  v8f;

#define B_  64
#define S_  2048
#define I_  256
#define H_  256
#define LDW 264   // padded LDS row stride (bf16); 528B = 132 dwords ≡ 4 mod 64 banks

static __device__ __forceinline__ v8f wmma_bf16(v16bf a, v16bf b, v8f c) {
    // v_wmma_f32_16x16x32_bf16: (neg_a, A, neg_b, B, c_mod, C, reuse_a, reuse_b)
    return __builtin_amdgcn_wmma_f32_16x16x32_bf16(false, a, false, b, (short)0, c, false, false);
}

static __device__ __forceinline__ float fast_tanh(float x) {
    // tanh(x) = 1 - 2/(exp(2x)+1); saturates correctly to +-1 for large |x|.
    float e = __builtin_amdgcn_exp2f(x * 2.885390081777927f);   // 2*log2(e)
    return __builtin_fmaf(-2.0f, __builtin_amdgcn_rcpf(e + 1.0f), 1.0f);
}

// A-matrix 16x32 bf16 fragment from row-major LDS tile (ISA 7.12.2):
// lane m(=lane&15) holds row m; lanes<16: K = kc+{0..7,16..23}; lanes>=16: kc+{8..15,24..31}.
static __device__ __forceinline__ v16bf load_A(const __bf16* lds, int mbase, int kc, int lane) {
    const int m   = mbase + (lane & 15);
    const int klo = kc + ((lane & 16) ? 8 : 0);
    const __bf16* p = lds + m * LDW + klo;
    v8bf lo = *(const v8bf*)(p);
    v8bf hi = *(const v8bf*)(p + 16);
    return __builtin_shufflevector(lo, hi, 0,1,2,3,4,5,6,7,8,9,10,11,12,13,14,15);
}

// B-matrix 32x16 bf16 fragment: lane n(=lane&15) holds column n;
// lanes<16: K = kc+{0..7,16..23}; lanes>=16: kc+{8..15,24..31}.
static __device__ __forceinline__ v16bf load_B(const __bf16* lds, int nbase, int kc, int lane) {
    const int n  = nbase + (lane & 15);
    const int kb = kc + ((lane & 16) ? 8 : 0);
    v16bf r;
    #pragma unroll
    for (int s = 0; s < 8; ++s) {
        r[s]     = lds[(kb + s) * LDW + n];
        r[s + 8] = lds[(kb + 16 + s) * LDW + n];
    }
    return r;
}

// ---------------- Kernel 1: input projection GEMM (M=B*S, N=H, K=I) ----------------
extern "C" __global__ __launch_bounds__(512)
void xh_gemm(const float* __restrict__ x, const float* __restrict__ W_xh,
             const float* __restrict__ b_xh, float* __restrict__ xp) {
    extern __shared__ char smem[];
    __bf16* Wt = (__bf16*)smem;                       // [I_][LDW]: Wt[k][n] = W_xh[n][k]
    __bf16* Ax = (__bf16*)(smem + I_ * LDW * 2);      // [64][LDW]
    const int tid  = threadIdx.x;
    const int row0 = blockIdx.x * 64;                 // row in flattened (B*S)

    for (int idx = tid; idx < H_ * I_; idx += 512) {
        int n = idx >> 8, k = idx & 255;
        Wt[k * LDW + n] = (__bf16)W_xh[idx];
    }
    for (int idx = tid; idx < 64 * I_; idx += 512) {
        int m = idx >> 8, k = idx & 255;
        Ax[m * LDW + k] = (__bf16)x[((row0 + m) << 8) + k];
    }
    __syncthreads();

    const int wave = tid >> 5, lane = tid & 31;
    const int rb  = wave >> 2;           // 0..3 : 16-row block
    const int cb0 = (wave & 3) * 4;      // 4 column tiles per wave
    v8f acc[4] = {};
    for (int kc = 0; kc < I_; kc += 32) {
        v16bf a = load_A(Ax, rb * 16, kc, lane);
        #pragma unroll
        for (int q = 0; q < 4; ++q) {
            v16bf b = load_B(Wt, (cb0 + q) * 16, kc, lane);
            acc[q] = wmma_bf16(a, b, acc[q]);
        }
    }
    const int n_lane = lane & 15;
    const int mo = rb * 16 + ((lane & 16) ? 8 : 0);
    #pragma unroll
    for (int q = 0; q < 4; ++q) {
        const int ncol = (cb0 + q) * 16 + n_lane;
        const float bias = b_xh[ncol];
        #pragma unroll
        for (int r = 0; r < 8; ++r)
            xp[((row0 + mo + r) << 8) + ncol] = acc[q][r] + bias;
    }
}

// ---------------- Kernel 2: persistent single-WGP recurrence ----------------
// 512 threads = 16 waves; each wave owns a 2x2 block of 16x16 tiles of h (64x256).
// W_hh B-fragments live in 128 VGPRs per wave for all 2048 steps.
extern "C" __global__ __launch_bounds__(512)
void rnn_recurrence(const float* __restrict__ W_hh, const float* __restrict__ b_hh,
                    float* __restrict__ out /* h_seq @0, h_last @ B*S*H = 1<<25 */) {
    extern __shared__ char smem[];
    __bf16* Wb = (__bf16*)smem;                       // [H_][LDW]: Wb[k][n] = W_hh[n][k]
    __bf16* hb = (__bf16*)(smem + H_ * LDW * 2);      // [B_][LDW]: current h (bf16)
    const int tid = threadIdx.x;

    for (int idx = tid; idx < H_ * H_; idx += 512) {
        int n = idx >> 8, k = idx & 255;
        Wb[k * LDW + n] = (__bf16)W_hh[idx];
    }
    for (int idx = tid; idx < B_ * LDW; idx += 512) hb[idx] = (__bf16)0.0f;
    __syncthreads();

    const int wave = tid >> 5, lane = tid & 31;
    const int rb0 = (wave >> 3) * 2;      // row blocks rb0, rb0+1
    const int cb0 = (wave & 7) * 2;       // col blocks cb0, cb0+1
    const int n_lane = lane & 15;
    const int mo_off = (lane & 16) ? 8 : 0;

    int col[2]; float bias[2]; int rrow[2];
    #pragma unroll
    for (int i = 0; i < 2; ++i) {
        col[i]  = (cb0 + i) * 16 + n_lane;
        bias[i] = b_hh[col[i]];
        rrow[i] = (rb0 + i) * 16 + mo_off;
    }

    // Hoist all B fragments (W_hh^T columns) into registers: 2 cb x 8 chunks x 8 VGPRs.
    v16bf Bf[2][8];
    #pragma unroll
    for (int i = 0; i < 2; ++i)
        #pragma unroll
        for (int kc8 = 0; kc8 < 8; ++kc8)
            Bf[i][kc8] = load_B(Wb, (cb0 + i) * 16, kc8 * 32, lane);

    // Preload xt for t=0 (xp lives in out). idx = (b<<19) + (t<<8) + col.
    float xn[2][2][8];
    #pragma unroll
    for (int ri = 0; ri < 2; ++ri)
        #pragma unroll
        for (int ci = 0; ci < 2; ++ci)
            #pragma unroll
            for (int r = 0; r < 8; ++r)
                xn[ri][ci][r] = out[((rrow[ri] + r) << 19) + col[ci]];

    for (int t = 0; t < S_; ++t) {
        // Fold xt + bias into the WMMA C-input.
        v8f acc[2][2];
        #pragma unroll
        for (int ri = 0; ri < 2; ++ri)
            #pragma unroll
            for (int ci = 0; ci < 2; ++ci)
                #pragma unroll
                for (int r = 0; r < 8; ++r)
                    acc[ri][ci][r] = xn[ri][ci][r] + bias[ci];

        // Prefetch next step's xt; hides L2 latency under the WMMA phase.
        if (t + 1 < S_) {
            const int tb = (t + 1) << 8;
            #pragma unroll
            for (int ri = 0; ri < 2; ++ri)
                #pragma unroll
                for (int ci = 0; ci < 2; ++ci)
                    #pragma unroll
                    for (int r = 0; r < 8; ++r)
                        xn[ri][ci][r] = out[((rrow[ri] + r) << 19) + tb + col[ci]];
        }

        // h @ W_hh^T : only A (h) comes from LDS now.
        #pragma unroll
        for (int kc8 = 0; kc8 < 8; ++kc8) {
            v16bf a0 = load_A(hb, rb0 * 16,       kc8 * 32, lane);
            v16bf a1 = load_A(hb, (rb0 + 1) * 16, kc8 * 32, lane);
            acc[0][0] = wmma_bf16(a0, Bf[0][kc8], acc[0][0]);
            acc[0][1] = wmma_bf16(a0, Bf[1][kc8], acc[0][1]);
            acc[1][0] = wmma_bf16(a1, Bf[0][kc8], acc[1][0]);
            acc[1][1] = wmma_bf16(a1, Bf[1][kc8], acc[1][1]);
        }

        __syncthreads();   // all hb reads for step t complete before overwrite
        const int tb = t << 8;
        #pragma unroll
        for (int ri = 0; ri < 2; ++ri)
            #pragma unroll
            for (int ci = 0; ci < 2; ++ci)
                #pragma unroll
                for (int r = 0; r < 8; ++r) {
                    float v = fast_tanh(acc[ri][ci][r]);
                    out[((rrow[ri] + r) << 19) + tb + col[ci]] = v;   // h_seq[b,t,:]
                    hb[(rrow[ri] + r) * LDW + col[ci]] = (__bf16)v;
                    if (t == S_ - 1)
                        out[(1 << 25) + ((rrow[ri] + r) << 8) + col[ci]] = v;  // h_last
                }
        __syncthreads();   // new h visible to all waves for step t+1
    }
}

extern "C" void kernel_launch(void* const* d_in, const int* in_sizes, int n_in,
                              void* d_out, int out_size, void* d_ws, size_t ws_size,
                              hipStream_t stream) {
    (void)in_sizes; (void)n_in; (void)d_ws; (void)ws_size; (void)out_size;
    const float* x    = (const float*)d_in[0];
    const float* W_xh = (const float*)d_in[1];
    const float* b_xh = (const float*)d_in[2];
    const float* W_hh = (const float*)d_in[3];
    const float* b_hh = (const float*)d_in[4];
    float* out = (float*)d_out;

    const size_t shmem_gemm = (size_t)(I_ * LDW + 64 * LDW) * sizeof(__bf16);  // ~165 KB
    const size_t shmem_rec  = (size_t)(H_ * LDW + B_ * LDW) * sizeof(__bf16);  // ~165 KB

    // xp -> h_seq slot of d_out (in place), whole-chip WMMA GEMM
    xh_gemm<<<dim3((B_ * S_) / 64), dim3(512), shmem_gemm, stream>>>(x, W_xh, b_xh, out);
    // serial recurrence on one WGP, reads xp from d_out and overwrites with h
    rnn_recurrence<<<dim3(1), dim3(512), shmem_rec, stream>>>(W_hh, b_hh, out);
}
